// TransformCTRNN2_45896020525284
// MI455X (gfx1250) — matrix operand order repaired and checked by reading, not compiled, not run on hardware
//
#include <hip/hip_runtime.h>
#include <hip/hip_bf16.h>

typedef __attribute__((ext_vector_type(16))) __bf16 v16bf;
typedef __attribute__((ext_vector_type(8)))  float  v8f;

#define SEQ   128
#define NB    32      // batch
#define NI    128     // input size
#define NH    256     // hidden
#define ALPHA_F 0.2f
#define OMA_F   0.8f

// ---------------- workspace layout (bytes) ----------------
// Asw : swizzled bf16 transform       67,108,864
// th  : fp32 EMA state [B][H][H]       8,388,608
// h   : fp32 hidden    [B][H]             32,768
// ip  : fp32 input_proj[B][H]             32,768
// Bsw : swizzled bf16 combined            32,768
// Pbuf: fp32 partials [H][16][B]         524,288
#define OFF_ASW   ((size_t)0)
#define OFF_TH    ((size_t)67108864)
#define OFF_H     (OFF_TH  + 8388608)
#define OFF_IP    (OFF_H   + 32768)
#define OFF_BSW   (OFF_IP  + 32768)
#define OFF_PBUF  (OFF_BSW + 32768)

__device__ __forceinline__ unsigned short f32_bf16(float f) {
  union { float f; unsigned u; } v; v.f = f;
  unsigned u = v.u;
  u += 0x7FFFu + ((u >> 16) & 1u);   // round-to-nearest-even
  return (unsigned short)(u >> 16);
}

// ---- one-time: fp32 transform [65536][512] -> bf16, swizzled into WMMA A lane order
// A 16-bit 16x32 layout (ISA table): lane half h, VGPR r holds K pair:
//   K = (r<4 ? 2r + 8h : 16 + 2(r-4) + 8h) + e
// storage: tile(mt,kb) = 1024B; per lane 32B contiguous (16 bf16 in (r,e) order)
__global__ __launch_bounds__(256) void k_convert_transform(
    const float* __restrict__ T, unsigned* __restrict__ Asw) {
  unsigned idx = blockIdx.x * 256u + threadIdx.x;   // dword index, 16,777,216 total
  unsigned s    = idx << 1;                         // bf16 element index (even)
  unsigned tile = s >> 9;                           // 512 bf16 per tile
  unsigned w    = s & 511u;
  unsigned lane = w >> 4;
  unsigned r    = (w & 15u) >> 1;                   // e = 0 (pair handled together)
  unsigned mt   = tile >> 4, kb = tile & 15u;
  unsigned m    = mt * 16u + (lane & 15u);
  unsigned hh   = lane >> 4;
  unsigned kbase = kb * 32u + (r < 4u ? 2u*r + 8u*hh : 16u + 2u*(r-4u) + 8u*hh);
  const float* src = T + (size_t)m * 512u + kbase;
  unsigned lo = f32_bf16(src[0]);
  unsigned hi = f32_bf16(src[1]);
  Asw[idx] = lo | (hi << 16);
}

__global__ __launch_bounds__(256) void k_init_th(float* __restrict__ th) {
  unsigned idx = blockIdx.x * 256u + threadIdx.x;   // 2,097,152 floats
  unsigned i = (idx >> 8) & 255u, j = idx & 255u;
  th[idx] = (i == j) ? 1.0f : 0.0f;
}

// ---- B operand scatter: element (kglobal, n) of combined^T [512 x 32]
// B 16-bit 32x16 layout: lane = h*16 + n%16 ; h = (kk>>3)&1 ; VGPR r = ((kk>>4)<<2) + ((kk&7)>>1) ; e = kk&1
__device__ __forceinline__ void scatterB(unsigned short* __restrict__ Bsw,
                                         int kglobal, int n, unsigned short v) {
  int kb = kglobal >> 5, kk = kglobal & 31;
  int hh = (kk >> 3) & 1;
  int r  = ((kk >> 4) << 2) + ((kk & 7) >> 1);
  int e  = kk & 1;
  int lane  = hh * 16 + (n & 15);
  int bhalf = n >> 4;
  Bsw[((((bhalf * 16 + kb) * 32) + lane) << 4) + r * 2 + e] = v;
}

// ---- per-step prep: input_proj + swizzled bf16 combined
__global__ __launch_bounds__(256) void k_prep(
    const float* __restrict__ x, const float* __restrict__ w_in,
    const float* __restrict__ b_in, const float* __restrict__ h,
    float* __restrict__ ip, unsigned short* __restrict__ Bsw, int t) {
  int b = blockIdx.x, i = threadIdx.x;
  const float* xr = x + ((size_t)t * NB + b) * NI;
  const float* wr = w_in + (size_t)i * NI;
  float acc = b_in[i];
#pragma unroll 8
  for (int k = 0; k < NI; ++k) acc += xr[k] * wr[k];
  ip[b * NH + i] = acc;
  float hv = h[b * NH + i];
  scatterB(Bsw, i,        b, f32_bf16(acc));
  scatterB(Bsw, NH + i,   b, f32_bf16(hv));
}

// ---- fused GEMM epilogue: th EMA + partial batched matvec
__device__ __forceinline__ void epilogue(
    v8f c, int b, int i, int jb, int jt, int lane,
    float* __restrict__ th, const float* __restrict__ ip, float* __restrict__ Pbuf) {
  size_t base = ((size_t)b * NH + i) * NH + jb;
  float4* thp = reinterpret_cast<float4*>(th + base);
  float4 t0 = thp[0], t1 = thp[1];
  const float4* ipp = reinterpret_cast<const float4*>(ip + b * NH + jb);
  float4 p0 = ipp[0], p1 = ipp[1];
  float n0 = t0.x * OMA_F + c[0] * ALPHA_F;
  float n1 = t0.y * OMA_F + c[1] * ALPHA_F;
  float n2 = t0.z * OMA_F + c[2] * ALPHA_F;
  float n3 = t0.w * OMA_F + c[3] * ALPHA_F;
  float n4 = t1.x * OMA_F + c[4] * ALPHA_F;
  float n5 = t1.y * OMA_F + c[5] * ALPHA_F;
  float n6 = t1.z * OMA_F + c[6] * ALPHA_F;
  float n7 = t1.w * OMA_F + c[7] * ALPHA_F;
  thp[0] = make_float4(n0, n1, n2, n3);
  thp[1] = make_float4(n4, n5, n6, n7);
  float partial = n0*p0.x + n1*p0.y + n2*p0.z + n3*p0.w
                + n4*p1.x + n5*p1.y + n6*p1.z + n7*p1.w;
  partial += __shfl_xor(partial, 16);               // join j halves (same b)
  if (lane < 16) Pbuf[(i * 16 + jt) * NB + b] = partial;
}

// ---- big kernel: tm = transform @ combined^T (M=65536, N=32, K=512), fused EMA+matvec
// one wave per 16-row m-tile (fixed i, 16 consecutive j); 8 waves / block; 512 blocks
__global__ __launch_bounds__(256) void k_big(
    const v16bf* __restrict__ Asw, const v16bf* __restrict__ Bsw,
    float* __restrict__ th, const float* __restrict__ ip, float* __restrict__ Pbuf) {
  int lane = threadIdx.x & 31;
  int wave = threadIdx.x >> 5;
  int mt = blockIdx.x * 8 + wave;          // 0..4095
  int i  = mt >> 4;
  int jt = mt & 15;
  int j0 = jt << 4;

  v8f c0 = {0.f,0.f,0.f,0.f,0.f,0.f,0.f,0.f};
  v8f c1 = {0.f,0.f,0.f,0.f,0.f,0.f,0.f,0.f};
#pragma unroll
  for (int kb = 0; kb < 16; ++kb) {
    v16bf a  = Asw[(size_t)(mt * 16 + kb) * 32 + lane];
    v16bf b0 = Bsw[(kb)      * 32 + lane];
    v16bf b1 = Bsw[(16 + kb) * 32 + lane];
    c0 = __builtin_amdgcn_wmma_f32_16x16x32_bf16(false, a, false, b0, (short)0, c0, false, false);
    c1 = __builtin_amdgcn_wmma_f32_16x16x32_bf16(false, a, false, b1, (short)0, c1, false, false);
  }
  int jb = j0 + ((lane >> 4) << 3);        // +8 for lane half 1
  epilogue(c0, (lane & 15),      i, jb, jt, lane, th, ip, Pbuf);
  epilogue(c1, 16 + (lane & 15), i, jb, jt, lane, th, ip, Pbuf);
}

// ---- finalize: transformed + h@w_h.T + b_h, relu, h EMA, emit output[t]
__global__ __launch_bounds__(256) void k_finalize(
    const float* __restrict__ w_h, const float* __restrict__ b_h,
    const float* __restrict__ Pbuf, float* __restrict__ h,
    float* __restrict__ out, int t) {
  __shared__ float hsh[NH];
  int b = blockIdx.x, i = threadIdx.x;
  hsh[i] = h[b * NH + i];
  __syncthreads();
  float acc = b_h[i];
  const float* wr = w_h + (size_t)i * NH;
#pragma unroll 8
  for (int k = 0; k < NH; ++k) acc += hsh[k] * wr[k];
  float tr = 0.f;
#pragma unroll
  for (int jt = 0; jt < 16; ++jt) tr += Pbuf[(i * 16 + jt) * NB + b];
  float hn = tr + acc;
  hn = hn > 0.f ? hn : 0.f;
  float hu = hsh[i] * OMA_F + hn * ALPHA_F;
  h[b * NH + i] = hu;
  out[(size_t)t * (NB * NH) + b * NH + i] = hu;
}

extern "C" void kernel_launch(void* const* d_in, const int* in_sizes, int n_in,
                              void* d_out, int out_size, void* d_ws, size_t ws_size,
                              hipStream_t stream) {
  (void)in_sizes; (void)n_in; (void)out_size; (void)ws_size;
  const float* x         = (const float*)d_in[0];  // [128][32][128]
  const float* hidden    = (const float*)d_in[1];  // [32][256]
  const float* w_in      = (const float*)d_in[2];  // [256][128]
  const float* b_in      = (const float*)d_in[3];  // [256]
  const float* w_h       = (const float*)d_in[4];  // [256][256]
  const float* b_h       = (const float*)d_in[5];  // [256]
  const float* transform = (const float*)d_in[6];  // [65536][512]
  // d_in[7] = num_steps (static python int = 1 in reference)

  unsigned char* ws = (unsigned char*)d_ws;
  unsigned*       Asw  = (unsigned*)(ws + OFF_ASW);
  float*          th   = (float*)(ws + OFF_TH);
  float*          h    = (float*)(ws + OFF_H);
  float*          ip   = (float*)(ws + OFF_IP);
  unsigned short* Bsw  = (unsigned short*)(ws + OFF_BSW);
  float*          Pbuf = (float*)(ws + OFF_PBUF);
  float* out = (float*)d_out;

  hipMemcpyAsync(h, hidden, NB * NH * sizeof(float), hipMemcpyDeviceToDevice, stream);
  k_convert_transform<<<65536, 256, 0, stream>>>(transform, Asw);
  k_init_th<<<8192, 256, 0, stream>>>(th);

  for (int t = 0; t < SEQ; ++t) {
    k_prep<<<NB, NH, 0, stream>>>(x, w_in, b_in, h, ip, Bsw, t);
    k_big<<<512, 256, 0, stream>>>((const v16bf*)Asw, (const v16bf*)Bsw, th, ip, Pbuf);
    k_finalize<<<NB, NH, 0, stream>>>(w_h, b_h, Pbuf, h, out, t);
  }
  hipMemcpyAsync(out + (size_t)SEQ * NB * NH, h, NB * NH * sizeof(float),
                 hipMemcpyDeviceToDevice, stream);
}
